// DNATransportGNN_55619826483375
// MI455X (gfx1250) — compile-verified, matrix operand order checked by reading, not run on hardware
//
#include <hip/hip_runtime.h>
#include <hip/hip_bf16.h>
#include <math.h>

// ---------------- problem constants ----------------
#define N_NODES 40000
#define N_EDGES 640000
#define N_GRAPH 32
#define HDIM    128
#define HEADS   4
#define LAYERS  4
#define ODIM    100
#define CDIM    32          // HDIM / HEADS

typedef __bf16 bf16;
typedef __attribute__((ext_vector_type(16))) __bf16 v16bf;
typedef __attribute__((ext_vector_type(8)))  __bf16 v8bf;
typedef __attribute__((ext_vector_type(8)))  float  v8f;

union Frag16 { v16bf v; v8bf h[2]; };

// ---------------- CDNA5 async global->LDS helpers ----------------
// GLOBAL_LOAD_ASYNC_TO_LDS_B128: copies global memory into LDS, 16 bytes
// per lane, tracked with ASYNCcnt.
__device__ inline void async_load16_to_lds(unsigned lds_addr, const void* gaddr) {
    asm volatile("global_load_async_to_lds_b128 %0, %1, off"
                 :: "v"(lds_addr), "v"(gaddr) : "memory");
}
__device__ inline void wait_asynccnt0() {
#if __has_builtin(__builtin_amdgcn_s_wait_asynccnt)
    __builtin_amdgcn_s_wait_asynccnt(0);
#else
    asm volatile("s_wait_asynccnt 0x0" ::: "memory");
#endif
}

// ---------------- small utility kernels ----------------
__global__ void fill_f32(float* __restrict__ p, float v, int n) {
    int i = blockIdx.x * blockDim.x + threadIdx.x;
    if (i < n) p[i] = v;
}

// h = x @ node_w + node_b  (K = 4), also emit bf16 copy for WMMA
__global__ void node_proj(const float* __restrict__ x, const float* __restrict__ w,
                          const float* __restrict__ b, float* __restrict__ h,
                          bf16* __restrict__ hb) {
    int i = blockIdx.x * blockDim.x + threadIdx.x;
    if (i >= N_NODES * HDIM) return;
    int n = i >> 7, c = i & 127;
    const float* xr = x + (size_t)n * 4;
    float acc = b[c]
              + xr[0] * w[0 * HDIM + c] + xr[1] * w[1 * HDIM + c]
              + xr[2] * w[2 * HDIM + c] + xr[3] * w[3 * HDIM + c];
    h[i]  = acc;
    hb[i] = (bf16)acc;
}

// e_h = edge_attr @ edge_w + edge_b (K = 5), bf16 only (it is GEMM input only)
__global__ void edge_proj(const float* __restrict__ ea, const float* __restrict__ w,
                          const float* __restrict__ b, bf16* __restrict__ ehb) {
    int i = blockIdx.x * blockDim.x + threadIdx.x;
    if (i >= N_EDGES * HDIM) return;
    int e = i >> 7, c = i & 127;
    const float* er = ea + (size_t)e * 5;
    float acc = b[c];
#pragma unroll
    for (int j = 0; j < 5; ++j) acc += er[j] * w[j * HDIM + c];
    ehb[i] = (bf16)acc;
}

// Wt[n,k] = (bf16) W[k,n]  (128x128) -> B-panel friendly layout for WMMA
__global__ void transpose_w(const float* __restrict__ W, bf16* __restrict__ Wt) {
    int i = blockIdx.x * blockDim.x + threadIdx.x;
    if (i >= HDIM * HDIM) return;
    int k = i >> 7, n = i & 127;
    Wt[n * HDIM + k] = (bf16)W[i];
}

// ---------------- WMMA GEMM: C[M,128] = A_bf16[M,128] @ W[128,128] (+bias) ----
// Block = 64 rows x 128 cols, 256 threads = 8 waves; wave w owns column tile
// n0 = 16*w and iterates 4 row tiles. A strip (64x128 bf16) is staged once into
// LDS via GLOBAL_LOAD_ASYNC_TO_LDS_B128 (ASYNCcnt), removing the 8x redundant
// global A traffic; fragments then come from conflict-free ds_load_b128
// (row stride padded to 136 bf16 = +4 banks/row). Per K-chunk all four A
// fragments are loaded before the four WMMAs so DS waits stay partial and
// overlap WMMA execution. fp32 accumulation; EXEC all-1s around WMMA.
#define LDS_STRIDE 136      // bf16 elements per LDS row (128 + 8 pad)

__global__ __launch_bounds__(256)
void wmma_gemm128(const bf16* __restrict__ A, const bf16* __restrict__ Wt,
                  const float* __restrict__ bias, float* __restrict__ C) {
    __shared__ bf16 sA[64 * LDS_STRIDE];        // 17408 bytes

    const int tid  = threadIdx.x;
    const int lane = tid & 31;
    const int wave = tid >> 5;
    const int m0   = blockIdx.x << 6;           // 64-row strip
    const int n0   = wave << 4;
    const int col  = lane & 15;
    const int half = lane >> 4;
    const int koff = half << 3;                 // lanes 16-31 start at K+8

    // ---- stage A strip into LDS: 64 rows x 256B = 1024 x 16B chunks ----
#pragma unroll
    for (int it = 0; it < 4; ++it) {
        int t   = tid + it * 256;               // chunk id 0..1023
        int row = t >> 4;                       // 0..63
        int chk = t & 15;                       // 16B chunk within row
        const bf16* gsrc = A + (size_t)(m0 + row) * HDIM + chk * 8;
        unsigned lds_addr =
            (unsigned)(unsigned long long)&sA[row * LDS_STRIDE + chk * 8];
        async_load16_to_lds(lds_addr, gsrc);
    }
    wait_asynccnt0();
    __syncthreads();

    // ---- compute ----
    const bf16* brow = Wt + (size_t)(n0 + col) * HDIM + koff;
    const bf16* aLds = sA + (size_t)col * LDS_STRIDE + koff;

    v8f acc[4] = {v8f{}, v8f{}, v8f{}, v8f{}};
#pragma unroll
    for (int kk = 0; kk < HDIM; kk += 32) {
        Frag16 b;
        b.h[0] = *(const v8bf*)(brow + kk);
        b.h[1] = *(const v8bf*)(brow + kk + 16);
        // batch all 4 row-tile fragments first -> partial dscnt waits
        Frag16 a[4];
#pragma unroll
        for (int mt = 0; mt < 4; ++mt) {
            const bf16* ap = aLds + mt * 16 * LDS_STRIDE + kk;
            a[mt].h[0] = *(const v8bf*)(ap);
            a[mt].h[1] = *(const v8bf*)(ap + 16);
        }
#pragma unroll
        for (int mt = 0; mt < 4; ++mt)
            acc[mt] = __builtin_amdgcn_wmma_f32_16x16x32_bf16(
                          false, a[mt].v, false, b.v, (short)0, acc[mt], false, false);
    }

    const int n = n0 + col;
    const float bv = bias ? bias[n] : 0.0f;
#pragma unroll
    for (int mt = 0; mt < 4; ++mt) {
        float* crow = C + (size_t)(m0 + mt * 16 + half * 8) * HDIM + n;
#pragma unroll
        for (int r = 0; r < 8; ++r)
            crow[(size_t)r * HDIM] = acc[mt][r] + bv;
    }
}

// ---------------- scatter softmax (3 passes over edges) ----------------
__device__ inline void atomicMaxF(float* addr, float val) {
    if (__float_as_int(val) >= 0)
        atomicMax((int*)addr, __float_as_int(val));
    else
        atomicMin((unsigned int*)addr, (unsigned int)__float_as_int(val));
}

// pass 1: logits alpha[e,h] = scale * dot(q[dst], k[src]+e) ; running max per (dst,h)
__global__ void alpha_kernel(const float* __restrict__ q, const float* __restrict__ k,
                             const float* __restrict__ e, const int* __restrict__ ei,
                             float* __restrict__ alpha, float* __restrict__ mmax) {
    int i = blockIdx.x * blockDim.x + threadIdx.x;
    if (i >= N_EDGES * HEADS) return;
    int eid = i >> 2, hh = i & 3;
    int src = ei[eid], dst = ei[N_EDGES + eid];
    const float4* qr = (const float4*)(q + (size_t)dst * HDIM + hh * CDIM);
    const float4* kr = (const float4*)(k + (size_t)src * HDIM + hh * CDIM);
    const float4* er = (const float4*)(e + (size_t)eid * HDIM + hh * CDIM);
    float s = 0.f;
#pragma unroll
    for (int j = 0; j < CDIM / 4; ++j) {
        float4 qv = qr[j], kv = kr[j], ev = er[j];
        s += qv.x * (kv.x + ev.x) + qv.y * (kv.y + ev.y)
           + qv.z * (kv.z + ev.z) + qv.w * (kv.w + ev.w);
    }
    s *= 0.17677669529663687f;   // 1/sqrt(C)
    alpha[i] = s;
    atomicMaxF(&mmax[dst * HEADS + hh], s);
}

// pass 2: alpha <- exp(alpha - max) (in place); running sum per (dst,h)
__global__ void expsum_kernel(const int* __restrict__ ei, float* __restrict__ alpha,
                              const float* __restrict__ mmax, float* __restrict__ ssum) {
    int i = blockIdx.x * blockDim.x + threadIdx.x;
    if (i >= N_EDGES * HEADS) return;
    int eid = i >> 2, hh = i & 3;
    int dst = ei[N_EDGES + eid];
    float w = __expf(alpha[i] - mmax[dst * HEADS + hh]);
    alpha[i] = w;
    atomicAdd(&ssum[dst * HEADS + hh], w);
}

// pass 3: out[dst] += (v[src]+e) * alpha/(sum+eps)
__global__ void scatter_kernel(const int* __restrict__ ei, const float* __restrict__ alpha,
                               const float* __restrict__ ssum, const float* __restrict__ v,
                               const float* __restrict__ e, float* __restrict__ outacc) {
    int i = blockIdx.x * blockDim.x + threadIdx.x;
    if (i >= N_EDGES * HDIM) return;
    int eid = i >> 7, c = i & 127, hh = c >> 5;
    int src = ei[eid], dst = ei[N_EDGES + eid];
    float coeff = alpha[eid * HEADS + hh] / (ssum[dst * HEADS + hh] + 1e-16f);
    float msg = (v[(size_t)src * HDIM + c] + e[(size_t)eid * HDIM + c]) * coeff;
    atomicAdd(&outacc[(size_t)dst * HDIM + c], msg);
}

// ---------------- LayerNorm + ReLU (one wave32 per node, H=128) -------------
__global__ void ln_relu_kernel(const float* __restrict__ outacc, const float* __restrict__ skip,
                               const float* __restrict__ g, const float* __restrict__ b,
                               float* __restrict__ h, bf16* __restrict__ hb) {
    int node = (blockIdx.x * blockDim.x + threadIdx.x) >> 5;
    int lane = threadIdx.x & 31;
    if (node >= N_NODES) return;
    size_t base = (size_t)node * HDIM + lane * 4;
    float4 x  = *(const float4*)(outacc + base);
    float4 sk = *(const float4*)(skip + base);
    x.x += sk.x; x.y += sk.y; x.z += sk.z; x.w += sk.w;

    float s = x.x + x.y + x.z + x.w;
#pragma unroll
    for (int m = 16; m > 0; m >>= 1) s += __shfl_xor(s, m, 32);
    float mu = s * (1.0f / HDIM);
    float4 d = { x.x - mu, x.y - mu, x.z - mu, x.w - mu };
    float vs = d.x * d.x + d.y * d.y + d.z * d.z + d.w * d.w;
#pragma unroll
    for (int m = 16; m > 0; m >>= 1) vs += __shfl_xor(vs, m, 32);
    float inv = rsqrtf(vs * (1.0f / HDIM) + 1e-5f);

    float4 gg = *(const float4*)(g + lane * 4);
    float4 bb = *(const float4*)(b + lane * 4);
    float4 y;
    y.x = fmaxf(d.x * inv * gg.x + bb.x, 0.f);
    y.y = fmaxf(d.y * inv * gg.y + bb.y, 0.f);
    y.z = fmaxf(d.z * inv * gg.z + bb.z, 0.f);
    y.w = fmaxf(d.w * inv * gg.w + bb.w, 0.f);
    *(float4*)(h + base) = y;
    bf16* hp = hb + base;
    hp[0] = (bf16)y.x; hp[1] = (bf16)y.y; hp[2] = (bf16)y.z; hp[3] = (bf16)y.w;
}

// ---------------- pooling + heads ----------------
__global__ void pool_kernel(const float* __restrict__ h, const int* __restrict__ batch,
                            float* __restrict__ sums, float* __restrict__ cnt) {
    int i = blockIdx.x * blockDim.x + threadIdx.x;
    if (i >= N_NODES * HDIM) return;
    int n = i >> 7, c = i & 127;
    int g = batch[n];
    atomicAdd(&sums[g * HDIM + c], h[i]);
    if (c == 0) atomicAdd(&cnt[g], 1.0f);
}

__global__ void hg_kernel(const float* __restrict__ sums, const float* __restrict__ cnt,
                          float* __restrict__ hg) {
    int i = blockIdx.x * blockDim.x + threadIdx.x;
    if (i >= N_GRAPH * HDIM) return;
    hg[i] = sums[i] / fmaxf(cnt[i >> 7], 1.0f);
}

// both MLP heads; grid = G graphs, 128 threads
__global__ void head_mlp(const float* __restrict__ hg,
                         const float* __restrict__ w1a, const float* __restrict__ b1a,
                         const float* __restrict__ w2a, const float* __restrict__ b2a,
                         const float* __restrict__ w1b, const float* __restrict__ b1b,
                         const float* __restrict__ w2b, const float* __restrict__ b2b,
                         float* __restrict__ out) {
    __shared__ float sh[HDIM];
    __shared__ float st[HDIM / 2];
    int g = blockIdx.x, t = threadIdx.x;
    if (t < HDIM) sh[t] = hg[g * HDIM + t];
    __syncthreads();
    if (t < HDIM / 2) {
        float s = b1a[t];
        for (int j = 0; j < HDIM; ++j) s += sh[j] * w1a[j * (HDIM / 2) + t];
        st[t] = fmaxf(s, 0.f);
    }
    __syncthreads();
    if (t < ODIM) {
        float s = b2a[t];
        for (int j = 0; j < HDIM / 2; ++j) s += st[j] * w2a[j * ODIM + t];
        out[g * ODIM + t] = s;
    }
    __syncthreads();
    if (t < HDIM / 2) {
        float s = b1b[t];
        for (int j = 0; j < HDIM; ++j) s += sh[j] * w1b[j * (HDIM / 2) + t];
        st[t] = fmaxf(s, 0.f);
    }
    __syncthreads();
    if (t < ODIM) {
        float s = b2b[t];
        for (int j = 0; j < HDIM / 2; ++j) s += st[j] * w2b[j * ODIM + t];
        out[N_GRAPH * ODIM + g * ODIM + t] = s;
    }
}

// ---------------- host orchestration ----------------
extern "C" void kernel_launch(void* const* d_in, const int* in_sizes, int n_in,
                              void* d_out, int out_size, void* d_ws, size_t ws_size,
                              hipStream_t stream) {
    (void)in_sizes; (void)n_in; (void)out_size; (void)ws_size;
    const float* x         = (const float*)d_in[0];
    const float* edge_attr = (const float*)d_in[1];
    const float* node_w    = (const float*)d_in[2];
    const float* node_b    = (const float*)d_in[3];
    const float* edge_w    = (const float*)d_in[4];
    const float* edge_b    = (const float*)d_in[5];
    const float* Wq = (const float*)d_in[6];   const float* bq = (const float*)d_in[7];
    const float* Wk = (const float*)d_in[8];   const float* bk = (const float*)d_in[9];
    const float* Wv = (const float*)d_in[10];  const float* bv = (const float*)d_in[11];
    const float* We = (const float*)d_in[12];
    const float* Wskip = (const float*)d_in[13]; const float* bskip = (const float*)d_in[14];
    const float* ln_g = (const float*)d_in[15];  const float* ln_b = (const float*)d_in[16];
    const float* dos_w1 = (const float*)d_in[17]; const float* dos_b1 = (const float*)d_in[18];
    const float* dos_w2 = (const float*)d_in[19]; const float* dos_b2 = (const float*)d_in[20];
    const float* tr_w1  = (const float*)d_in[21]; const float* tr_b1  = (const float*)d_in[22];
    const float* tr_w2  = (const float*)d_in[23]; const float* tr_b2  = (const float*)d_in[24];
    const int* edge_index = (const int*)d_in[25];
    const int* batch      = (const int*)d_in[26];
    float* out = (float*)d_out;

    // ---- workspace carve-out (256B aligned) ----
    char* base = (char*)d_ws;
    size_t off = 0;
    auto carve = [&](size_t bytes) {
        char* p = base + off;
        off += (bytes + 255) & ~(size_t)255;
        return p;
    };
    float* h_f32  = (float*)carve((size_t)N_NODES * HDIM * 4);
    bf16*  h_bf   = (bf16*) carve((size_t)N_NODES * HDIM * 2);
    bf16*  ehb    = (bf16*) carve((size_t)N_EDGES * HDIM * 2);
    float* qb     = (float*)carve((size_t)N_NODES * HDIM * 4);
    float* kb     = (float*)carve((size_t)N_NODES * HDIM * 4);
    float* vb     = (float*)carve((size_t)N_NODES * HDIM * 4);
    float* skipb  = (float*)carve((size_t)N_NODES * HDIM * 4);
    float* e_f32  = (float*)carve((size_t)N_EDGES * HDIM * 4);
    float* alphab = (float*)carve((size_t)N_EDGES * HEADS * 4);
    float* mmax   = (float*)carve((size_t)N_NODES * HEADS * 4);
    float* ssum   = (float*)carve((size_t)N_NODES * HEADS * 4);
    float* outacc = (float*)carve((size_t)N_NODES * HDIM * 4);
    bf16*  wt     = (bf16*) carve((size_t)5 * LAYERS * HDIM * HDIM * 2);
    float* psum   = (float*)carve((size_t)N_GRAPH * HDIM * 4);
    float* pcnt   = (float*)carve((size_t)N_GRAPH * 4);
    float* hgb    = (float*)carve((size_t)N_GRAPH * HDIM * 4);

    const int TB = 256;
    const int NH_BLK = (N_NODES * HDIM + TB - 1) / TB;   // 20000
    const int EH_BLK = (N_EDGES * HDIM + TB - 1) / TB;   // 320000
    const int E4_BLK = (N_EDGES * HEADS + TB - 1) / TB;  // 10000

    // input projections
    node_proj<<<NH_BLK, TB, 0, stream>>>(x, node_w, node_b, h_f32, h_bf);
    edge_proj<<<EH_BLK, TB, 0, stream>>>(edge_attr, edge_w, edge_b, ehb);

    // pre-transpose all weights to bf16 B-panels: slots {q,k,v,skip,e} x layers
    const float* wsrc[5] = { Wq, Wk, Wv, Wskip, We };
    for (int s = 0; s < 5; ++s)
        for (int l = 0; l < LAYERS; ++l)
            transpose_w<<<(HDIM * HDIM) / TB, TB, 0, stream>>>(
                wsrc[s] + (size_t)l * HDIM * HDIM,
                wt + ((size_t)(s * LAYERS + l)) * HDIM * HDIM);

    for (int i = 0; i < LAYERS; ++i) {
        bf16* wtq = wt + ((size_t)(0 * LAYERS + i)) * HDIM * HDIM;
        bf16* wtk = wt + ((size_t)(1 * LAYERS + i)) * HDIM * HDIM;
        bf16* wtv = wt + ((size_t)(2 * LAYERS + i)) * HDIM * HDIM;
        bf16* wts = wt + ((size_t)(3 * LAYERS + i)) * HDIM * HDIM;
        bf16* wte = wt + ((size_t)(4 * LAYERS + i)) * HDIM * HDIM;

        wmma_gemm128<<<N_NODES / 64, TB, 0, stream>>>(h_bf, wtq, bq + i * HDIM, qb);
        wmma_gemm128<<<N_NODES / 64, TB, 0, stream>>>(h_bf, wtk, bk + i * HDIM, kb);
        wmma_gemm128<<<N_NODES / 64, TB, 0, stream>>>(h_bf, wtv, bv + i * HDIM, vb);
        wmma_gemm128<<<N_NODES / 64, TB, 0, stream>>>(h_bf, wts, bskip + i * HDIM, skipb);
        wmma_gemm128<<<N_EDGES / 64, TB, 0, stream>>>(ehb, wte, nullptr, e_f32);

        fill_f32<<<(N_NODES * HEADS + TB - 1) / TB, TB, 0, stream>>>(mmax, -INFINITY, N_NODES * HEADS);
        fill_f32<<<(N_NODES * HEADS + TB - 1) / TB, TB, 0, stream>>>(ssum, 0.0f, N_NODES * HEADS);
        fill_f32<<<NH_BLK, TB, 0, stream>>>(outacc, 0.0f, N_NODES * HDIM);

        alpha_kernel  <<<E4_BLK, TB, 0, stream>>>(qb, kb, e_f32, edge_index, alphab, mmax);
        expsum_kernel <<<E4_BLK, TB, 0, stream>>>(edge_index, alphab, mmax, ssum);
        scatter_kernel<<<EH_BLK, TB, 0, stream>>>(edge_index, alphab, ssum, vb, e_f32, outacc);

        ln_relu_kernel<<<N_NODES / 8, TB, 0, stream>>>(outacc, skipb,
                                                       ln_g + i * HDIM, ln_b + i * HDIM,
                                                       h_f32, h_bf);
    }

    // global mean pool + heads
    fill_f32<<<(N_GRAPH * HDIM + TB - 1) / TB, TB, 0, stream>>>(psum, 0.0f, N_GRAPH * HDIM);
    fill_f32<<<1, TB, 0, stream>>>(pcnt, 0.0f, N_GRAPH);
    pool_kernel<<<NH_BLK, TB, 0, stream>>>(h_f32, batch, psum, pcnt);
    hg_kernel<<<(N_GRAPH * HDIM + TB - 1) / TB, TB, 0, stream>>>(psum, pcnt, hgb);
    head_mlp<<<N_GRAPH, HDIM, 0, stream>>>(hgb,
                                           dos_w1, dos_b1, dos_w2, dos_b2,
                                           tr_w1, tr_b1, tr_w2, tr_b2, out);
}
